// WindGNN_62706522521633
// MI455X (gfx1250) — matrix operand order repaired and checked by reading, not compiled, not run on hardware
//
#include <hip/hip_runtime.h>
#include <hip/hip_bf16.h>

// ---------------------------------------------------------------------------
// WindGNN on MI455X (gfx1250): bf16 WMMA message passing.
//   N=50000 nodes, E=600000 edges, LAT=HID=128, 8 MP layers.
// All matmuls via V_WMMA_F32_16X16X32_BF16 (fp32 accumulate).
// h, agg kept fp32 (L2-resident); ea streamed as bf16 (halves HBM traffic).
// Epilogues: v_rcp_f32/v_rsq_f32 for SiLU/LN, native fptrunc for f32->bf16.
// ---------------------------------------------------------------------------

#define GNN_N 50000
#define GNN_E 600000
#define EW 4   // waves per block (128 threads, wave32)

typedef __attribute__((ext_vector_type(16))) __bf16 v16bf;
typedef __attribute__((ext_vector_type(8)))  float  v8f;

union FragU {
    v16bf v;
    uint4 q[2];
};

// Native f32 -> bf16 (backend lowers fptrunc to the hardware convert).
__device__ __forceinline__ unsigned short f2bf(float f) {
    return __builtin_bit_cast(unsigned short, (__bf16)f);
}

__device__ __forceinline__ ushort4 f2bf4(float4 v) {
    ushort4 r;
    r.x = f2bf(v.x); r.y = f2bf(v.y); r.z = f2bf(v.z); r.w = f2bf(v.w);
    return r;
}

__device__ __forceinline__ float fast_rcp(float x) { return __builtin_amdgcn_rcpf(x); }
__device__ __forceinline__ float fast_rsq(float x) { return __builtin_amdgcn_rsqf(x); }

// SiLU with v_exp_f32 + v_rcp_f32 (avoids IEEE div expansion).
__device__ __forceinline__ float siluf(float x) {
    return x * fast_rcp(1.0f + __expf(-x));
}

// A: LDS tile, 16 rows x (KT*32) bf16, row stride = strideUS ushorts (mult of 8).
// Wf: global bf16 weight fragments laid out [kt][NTILES nt][32 lane][16 elem],
//     lane's 16 elems = K values of column nt*16+(lane&15) within the kt*32 chunk:
//     elems 0-7 -> K = (lane>>4)*8 + 0..7, 8-15 -> K = (lane>>4)*8 + 16..23 (ISA 16-bit B layout).
template<int KT, int NTILES>
__device__ __forceinline__ v8f gemm_acc(const unsigned short* A, int strideUS,
                                        const unsigned short* Wf, int nt,
                                        int lane, v8f acc) {
    const int m    = lane & 15;
    const int half = lane >> 4;
#pragma unroll
    for (int kt = 0; kt < KT; ++kt) {
        FragU a, b;
        // ISA 16-bit A layout: elems 0-7 -> K = half*8 + 0..7,
        //                      elems 8-15 -> K = half*8 + 16..23 (within kt*32)
        const unsigned short* ap = A + m * strideUS + kt * 32 + half * 8;
        a.q[0] = *reinterpret_cast<const uint4*>(ap);
        a.q[1] = *reinterpret_cast<const uint4*>(ap + 16);
        const unsigned short* bp = Wf + ((size_t)(kt * NTILES + nt) * 32 + lane) * 16;
        b.q[0] = *reinterpret_cast<const uint4*>(bp);
        b.q[1] = *reinterpret_cast<const uint4*>(bp + 8);
        acc = __builtin_amdgcn_wmma_f32_16x16x32_bf16(false, a.v, false, b.v,
                                                      (short)0, acc, false, false);
    }
    return acc;
}

// ---------------------------------------------------------------------------
// Weight fp32 [K, Ncols] -> bf16 fragment layout (K padded to mult of 32).
// ---------------------------------------------------------------------------
__global__ void wconv_kernel(const float* __restrict__ W,
                             unsigned short* __restrict__ dst,
                             int K, int Ncols) {
    const int i = blockIdx.x * blockDim.x + threadIdx.x;
    const int ktiles = (K + 31) >> 5;
    const int ntiles = Ncols >> 4;
    const int total = ktiles * ntiles * 512;
    if (i >= total) return;
    const int e    = i & 15;
    const int lane = (i >> 4) & 31;
    const int t    = i >> 9;
    const int nt   = t % ntiles;
    const int kt   = t / ntiles;
    const int k = (kt << 5) + ((lane >> 4) << 3) + (e & 7) + ((e >> 3) << 4);
    const int n = (nt << 4) + (lane & 15);
    const float v = (k < K) ? W[(size_t)k * Ncols + n] : 0.0f;
    dst[i] = f2bf(v);
}

__global__ void init_kernel(float* __restrict__ agg, float* __restrict__ cnt, int Nn) {
    const int i = blockIdx.x * blockDim.x + threadIdx.x;
    if (i < Nn * 128) agg[i] = 0.0f;
    if (i < Nn)       cnt[i] = 0.0f;
}

__global__ void count_kernel(const int* __restrict__ colI, float* __restrict__ cnt, int Ecnt) {
    const int i = blockIdx.x * blockDim.x + threadIdx.x;
    if (i < Ecnt) atomicAdd(&cnt[colI[i]], 1.0f);
}

// ---------------------------------------------------------------------------
// Encoder: [rows, din] -> MLP(din->128 SiLU ->128 SiLU) -> LN -> f32 or bf16.
// ---------------------------------------------------------------------------
__global__ void __launch_bounds__(128) enc_kernel(
    const float* __restrict__ in, int din,
    const unsigned short* __restrict__ W1f,   // 1 ktile (Kpad=32), 8 ntiles
    const unsigned short* __restrict__ W2f,   // 4 ktiles, 8 ntiles
    const float* __restrict__ b1, const float* __restrict__ b2,
    const float* __restrict__ gg, const float* __restrict__ bb,
    float* __restrict__ outF, unsigned short* __restrict__ outB, int rows) {
    __shared__ unsigned short sA[EW][16 * 32];
    __shared__ unsigned short sH[EW][16 * 128];
    __shared__ float          sF[EW][16 * 128];
    const int wave = threadIdx.x >> 5;
    const int lane = threadIdx.x & 31;
    const int m    = lane & 15;
    const int half = lane >> 4;
    const int rbase = (blockIdx.x * EW + wave) << 4;
    if (rbase >= rows) return;
    unsigned short* A = sA[wave];
    unsigned short* H = sH[wave];
    float*          Fx = sF[wave];

    if (lane < 16) {
        const float* src = in + (size_t)(rbase + lane) * din;
        for (int j = 0; j < 32; ++j)
            A[lane * 32 + j] = (j < din) ? f2bf(src[j]) : (unsigned short)0;
    }
    for (int nt = 0; nt < 8; ++nt) {
        v8f acc = {};
        acc = gemm_acc<1, 8>(A, 32, W1f, nt, lane, acc);
        const int c = (nt << 4) + m;
        const float bias = b1[c];
#pragma unroll
        for (int r = 0; r < 8; ++r)
            H[(r + half * 8) * 128 + c] = f2bf(siluf(acc[r] + bias));
    }
    for (int nt = 0; nt < 8; ++nt) {
        v8f acc = {};
        acc = gemm_acc<4, 8>(H, 128, W2f, nt, lane, acc);
        const int c = (nt << 4) + m;
        const float bias = b2[c];
#pragma unroll
        for (int r = 0; r < 8; ++r)
            Fx[(r + half * 8) * 128 + c] = siluf(acc[r] + bias);
    }
    // LayerNorm over 128 dims: lane pair (m, m+16) covers the row.
    const float* fp = Fx + m * 128 + half * 64;
    float sum = 0.0f, sq = 0.0f;
    for (int j = 0; j < 64; ++j) { const float v = fp[j]; sum += v; sq += v * v; }
    sum += __shfl_xor(sum, 16);
    sq  += __shfl_xor(sq, 16);
    const float mu   = sum * (1.0f / 128.0f);
    const float rstd = fast_rsq(sq * (1.0f / 128.0f) - mu * mu + 1e-5f);
    if (outF) {
        float* dst = outF + (size_t)(rbase + m) * 128 + half * 64;
        for (int j = 0; j < 64; ++j) {
            const int c = half * 64 + j;
            dst[j] = (fp[j] - mu) * rstd * gg[c] + bb[c];
        }
    } else {
        for (int j = 0; j < 64; ++j) {
            const int c = half * 64 + j;
            H[m * 128 + c] = f2bf((fp[j] - mu) * rstd * gg[c] + bb[c]);
        }
        unsigned short* dst = outB + (size_t)rbase * 128;
#pragma unroll
        for (int it = 0; it < 8; ++it) {           // coalesced b128 stores
            const int o = it * 256 + lane * 8;
            *reinterpret_cast<uint4*>(dst + o) = *reinterpret_cast<const uint4*>(H + o);
        }
    }
}

// ---------------------------------------------------------------------------
// Edge message: ne = MLP([h[row]|h[col]|ea]); ea <- ne; agg[col] += ne.
// ---------------------------------------------------------------------------
__global__ void __launch_bounds__(128) edge_mp_kernel(
    const float* __restrict__ h,
    unsigned short* __restrict__ ea,          // in/out bf16 [E,128]
    float* __restrict__ agg,                  // [N,128] scatter target
    const int* __restrict__ rowI, const int* __restrict__ colI,
    const unsigned short* __restrict__ W1f,   // K=384 -> 12 ktiles
    const unsigned short* __restrict__ W2f,   // K=128 -> 4 ktiles
    const float* __restrict__ b1, const float* __restrict__ b2,
    const float* __restrict__ gg, const float* __restrict__ bb, int Ecnt) {
    __shared__ unsigned short sA[EW][16 * 384];
    __shared__ unsigned short sH[EW][16 * 128];
    __shared__ float          sF[EW][16 * 128];
    const int wave = threadIdx.x >> 5;
    const int lane = threadIdx.x & 31;
    const int m    = lane & 15;
    const int half = lane >> 4;
    const int ebase = (blockIdx.x * EW + wave) << 4;
    if (ebase >= Ecnt) return;
    unsigned short* A = sA[wave];
    unsigned short* H = sH[wave];
    float*          Fx = sF[wave];

    // Stage concat([h[row], h[col], ea]) as bf16 in LDS (h rows are L2-resident).
    for (int r = 0; r < 16; ++r) {
        const int e  = ebase + r;
        const int rn = rowI[e];
        const int cn = colI[e];
        const float4 hv = *reinterpret_cast<const float4*>(h + (size_t)rn * 128 + lane * 4);
        const float4 cv = *reinterpret_cast<const float4*>(h + (size_t)cn * 128 + lane * 4);
        *reinterpret_cast<ushort4*>(A + r * 384 + lane * 4)       = f2bf4(hv);
        *reinterpret_cast<ushort4*>(A + r * 384 + 128 + lane * 4) = f2bf4(cv);
        const ushort4 ev = *reinterpret_cast<const ushort4*>(ea + (size_t)e * 128 + lane * 4);
        *reinterpret_cast<ushort4*>(A + r * 384 + 256 + lane * 4) = ev;
    }
    // GEMM1: [16x384] @ [384x128] + SiLU -> bf16 hidden
    for (int nt = 0; nt < 8; ++nt) {
        v8f acc = {};
        acc = gemm_acc<12, 8>(A, 384, W1f, nt, lane, acc);
        const int c = (nt << 4) + m;
        const float bias = b1[c];
#pragma unroll
        for (int r = 0; r < 8; ++r)
            H[(r + half * 8) * 128 + c] = f2bf(siluf(acc[r] + bias));
    }
    // GEMM2: [16x128] @ [128x128] + SiLU -> f32
    for (int nt = 0; nt < 8; ++nt) {
        v8f acc = {};
        acc = gemm_acc<4, 8>(H, 128, W2f, nt, lane, acc);
        const int c = (nt << 4) + m;
        const float bias = b2[c];
#pragma unroll
        for (int r = 0; r < 8; ++r)
            Fx[(r + half * 8) * 128 + c] = siluf(acc[r] + bias);
    }
    // LayerNorm + scatter (message aggregation) + bf16 edge feature
    {
        const float* fp = Fx + m * 128 + half * 64;
        float sum = 0.0f, sq = 0.0f;
        for (int j = 0; j < 64; ++j) { const float v = fp[j]; sum += v; sq += v * v; }
        sum += __shfl_xor(sum, 16);
        sq  += __shfl_xor(sq, 16);
        const float mu   = sum * (1.0f / 128.0f);
        const float rstd = fast_rsq(sq * (1.0f / 128.0f) - mu * mu + 1e-5f);
        const int e = ebase + m;
        float* aggRow = agg + (size_t)colI[e] * 128 + half * 64;
        for (int j = 0; j < 64; ++j) {
            const int c = half * 64 + j;
            const float v = (fp[j] - mu) * rstd * gg[c] + bb[c];
            H[m * 128 + c] = f2bf(v);
            atomicAdd(aggRow + j, v);
        }
    }
    // Coalesced write-back of new edge features (in place).
    {
        unsigned short* dst = ea + (size_t)ebase * 128;
#pragma unroll
        for (int it = 0; it < 8; ++it) {
            const int o = it * 256 + lane * 8;
            *reinterpret_cast<uint4*>(dst + o) = *reinterpret_cast<const uint4*>(H + o);
        }
    }
}

// ---------------------------------------------------------------------------
// Node update: h += MLP([h | agg/cnt]); also zeroes agg for the next layer.
// ---------------------------------------------------------------------------
__global__ void __launch_bounds__(128) node_mp_kernel(
    float* __restrict__ h,
    float* __restrict__ agg,
    const float* __restrict__ cnt,
    const unsigned short* __restrict__ W1f,   // K=256 -> 8 ktiles
    const unsigned short* __restrict__ W2f,   // K=128 -> 4 ktiles
    const float* __restrict__ b1, const float* __restrict__ b2,
    const float* __restrict__ gg, const float* __restrict__ bb, int Ncnt) {
    __shared__ unsigned short sA[EW][16 * 256];
    __shared__ unsigned short sH[EW][16 * 128];
    __shared__ float          sF[EW][16 * 128];
    const int wave = threadIdx.x >> 5;
    const int lane = threadIdx.x & 31;
    const int m    = lane & 15;
    const int half = lane >> 4;
    const int nbase = (blockIdx.x * EW + wave) << 4;
    if (nbase >= Ncnt) return;
    unsigned short* A = sA[wave];
    unsigned short* H = sH[wave];
    float*          Fx = sF[wave];

    for (int r = 0; r < 16; ++r) {
        const int n = nbase + r;
        const float4 hv = *reinterpret_cast<const float4*>(h + (size_t)n * 128 + lane * 4);
        *reinterpret_cast<ushort4*>(A + r * 256 + lane * 4) = f2bf4(hv);
        const float ic = fast_rcp(fmaxf(cnt[n], 1.0f));
        float4 av = *reinterpret_cast<float4*>(agg + (size_t)n * 128 + lane * 4);
        av.x *= ic; av.y *= ic; av.z *= ic; av.w *= ic;
        *reinterpret_cast<ushort4*>(A + r * 256 + 128 + lane * 4) = f2bf4(av);
        float4 z; z.x = z.y = z.z = z.w = 0.0f;
        *reinterpret_cast<float4*>(agg + (size_t)n * 128 + lane * 4) = z;  // ready for next layer
    }
    for (int nt = 0; nt < 8; ++nt) {
        v8f acc = {};
        acc = gemm_acc<8, 8>(A, 256, W1f, nt, lane, acc);
        const int c = (nt << 4) + m;
        const float bias = b1[c];
#pragma unroll
        for (int r = 0; r < 8; ++r)
            H[(r + half * 8) * 128 + c] = f2bf(siluf(acc[r] + bias));
    }
    for (int nt = 0; nt < 8; ++nt) {
        v8f acc = {};
        acc = gemm_acc<4, 8>(H, 128, W2f, nt, lane, acc);
        const int c = (nt << 4) + m;
        const float bias = b2[c];
#pragma unroll
        for (int r = 0; r < 8; ++r)
            Fx[(r + half * 8) * 128 + c] = siluf(acc[r] + bias);
    }
    // LayerNorm + fp32 residual
    const float* fp = Fx + m * 128 + half * 64;
    float sum = 0.0f, sq = 0.0f;
    for (int j = 0; j < 64; ++j) { const float v = fp[j]; sum += v; sq += v * v; }
    sum += __shfl_xor(sum, 16);
    sq  += __shfl_xor(sq, 16);
    const float mu   = sum * (1.0f / 128.0f);
    const float rstd = fast_rsq(sq * (1.0f / 128.0f) - mu * mu + 1e-5f);
    float* hr = h + (size_t)(nbase + m) * 128 + half * 64;
#pragma unroll
    for (int j4 = 0; j4 < 16; ++j4) {
        float4 cur = *reinterpret_cast<float4*>(hr + j4 * 4);
        const int c = half * 64 + j4 * 4;
        cur.x += (fp[j4 * 4 + 0] - mu) * rstd * gg[c + 0] + bb[c + 0];
        cur.y += (fp[j4 * 4 + 1] - mu) * rstd * gg[c + 1] + bb[c + 1];
        cur.z += (fp[j4 * 4 + 2] - mu) * rstd * gg[c + 2] + bb[c + 2];
        cur.w += (fp[j4 * 4 + 3] - mu) * rstd * gg[c + 3] + bb[c + 3];
        *reinterpret_cast<float4*>(hr + j4 * 4) = cur;
    }
}

// ---------------------------------------------------------------------------
// Decoders: out[:,0] = dec_p(h), out[:,1:4] = dec_U(h).  128->128->64->{1,3}
// ---------------------------------------------------------------------------
__global__ void __launch_bounds__(128) dec_kernel(
    const float* __restrict__ h,
    const unsigned short* __restrict__ W1pf, const unsigned short* __restrict__ W2pf,
    const float* __restrict__ b1p, const float* __restrict__ b2p,
    const float* __restrict__ W3p, const float* __restrict__ b3p,
    const unsigned short* __restrict__ W1uf, const unsigned short* __restrict__ W2uf,
    const float* __restrict__ b1u, const float* __restrict__ b2u,
    const float* __restrict__ W3u, const float* __restrict__ b3u,
    float* __restrict__ out, int Ncnt) {
    __shared__ unsigned short sA[EW][16 * 128];
    __shared__ unsigned short sH[EW][16 * 128];
    __shared__ float          sF[EW][16 * 64];
    const int wave = threadIdx.x >> 5;
    const int lane = threadIdx.x & 31;
    const int m    = lane & 15;
    const int half = lane >> 4;
    const int nbase = (blockIdx.x * EW + wave) << 4;
    if (nbase >= Ncnt) return;
    unsigned short* A = sA[wave];
    unsigned short* H = sH[wave];
    float*          Fx = sF[wave];

    for (int r = 0; r < 16; ++r) {
        const float4 hv = *reinterpret_cast<const float4*>(h + (size_t)(nbase + r) * 128 + lane * 4);
        *reinterpret_cast<ushort4*>(A + r * 128 + lane * 4) = f2bf4(hv);
    }
    const unsigned short* W1f[2] = {W1pf, W1uf};
    const unsigned short* W2f[2] = {W2pf, W2uf};
    const float* b1a[2] = {b1p, b1u};
    const float* b2a[2] = {b2p, b2u};
    const float* W3a[2] = {W3p, W3u};
    const float* b3a[2] = {b3p, b3u};
    const int ocols[2] = {1, 3};
    const int ooff[2]  = {0, 1};

    for (int d = 0; d < 2; ++d) {
        for (int nt = 0; nt < 8; ++nt) {
            v8f acc = {};
            acc = gemm_acc<4, 8>(A, 128, W1f[d], nt, lane, acc);
            const int c = (nt << 4) + m;
            const float bias = b1a[d][c];
#pragma unroll
            for (int r = 0; r < 8; ++r)
                H[(r + half * 8) * 128 + c] = f2bf(siluf(acc[r] + bias));
        }
        for (int nt = 0; nt < 4; ++nt) {
            v8f acc = {};
            acc = gemm_acc<4, 4>(H, 128, W2f[d], nt, lane, acc);
            const int c = (nt << 4) + m;
            const float bias = b2a[d][c];
#pragma unroll
            for (int r = 0; r < 8; ++r)
                Fx[(r + half * 8) * 64 + c] = siluf(acc[r] + bias);
        }
        for (int col = 0; col < ocols[d]; ++col) {
            float part = 0.0f;
            for (int j = 0; j < 32; ++j) {
                const int kk = half * 32 + j;
                part += Fx[m * 64 + kk] * W3a[d][kk * ocols[d] + col];
            }
            part += __shfl_xor(part, 16);
            if (half == 0)
                out[(size_t)(nbase + m) * 4 + ooff[d] + col] = part + b3a[d][col];
        }
    }
}

// ---------------------------------------------------------------------------
// Host launcher.
// ---------------------------------------------------------------------------
extern "C" void kernel_launch(void* const* d_in, const int* in_sizes, int n_in,
                              void* d_out, int out_size, void* d_ws, size_t ws_size,
                              hipStream_t stream) {
    (void)in_sizes; (void)n_in; (void)out_size; (void)ws_size;
    constexpr int N = GNN_N, E = GNN_E;
    auto F = [&](int i) { return (const float*)d_in[i]; };
    // JAX pytree (sorted dict keys):
    // 0:x 1:edge_attr
    // dec_U: 2:W1 3:W2 4:W3 5:b1 6:b2 7:b3
    // dec_p: 8:W1 9:W2 10:W3 11:b1 12:b2 13:b3
    // edge_enc: 14:W1 15:W2 16:b1 17:b2 18:be 19:g
    // layers i (base 20+i*12): eW1,eW2,eb1,eb2,ebe,eg,nW1,nW2,nb1,nb2,nbe,ng
    // node_enc: 116:W1 117:W2 118:b1 119:b2 120:be 121:g
    // 122: edge_index [2,E]
    const float* x     = F(0);
    const float* eattr = F(1);
    const int* eidx = (const int*)d_in[122];
    const int* rowI = eidx;
    const int* colI = eidx + E;
    float* out = (float*)d_out;

    char* ws = (char*)d_ws;
    size_t o = 0;
    auto take = [&](size_t bytes) -> char* {
        char* p = ws + o;
        o += (bytes + 255) & ~(size_t)255;
        return p;
    };
    float* h            = (float*)take((size_t)N * 128 * 4);
    float* agg          = (float*)take((size_t)N * 128 * 4);
    float* cnt          = (float*)take((size_t)N * 4);
    unsigned short* ea  = (unsigned short*)take((size_t)E * 128 * 2);

    auto frag = [&](int pidx, int K, int Ncols) -> unsigned short* {
        const int ktiles = (K + 31) >> 5;
        const int ntiles = Ncols >> 4;
        const int total = ktiles * ntiles * 512;
        unsigned short* dst = (unsigned short*)take((size_t)total * 2);
        wconv_kernel<<<(total + 255) / 256, 256, 0, stream>>>(F(pidx), dst, K, Ncols);
        return dst;
    };

    unsigned short* ne_W1 = frag(116, 8, 128);
    unsigned short* ne_W2 = frag(117, 128, 128);
    unsigned short* ee_W1 = frag(14, 4, 128);
    unsigned short* ee_W2 = frag(15, 128, 128);
    unsigned short *leW1[8], *leW2[8], *lnW1[8], *lnW2[8];
    for (int i = 0; i < 8; ++i) {
        const int b = 20 + i * 12;
        leW1[i] = frag(b + 0, 384, 128);
        leW2[i] = frag(b + 1, 128, 128);
        lnW1[i] = frag(b + 6, 256, 128);
        lnW2[i] = frag(b + 7, 128, 128);
    }
    unsigned short* dU_W1 = frag(2, 128, 128);
    unsigned short* dU_W2 = frag(3, 128, 64);
    unsigned short* dP_W1 = frag(8, 128, 128);
    unsigned short* dP_W2 = frag(9, 128, 64);

    init_kernel<<<(N * 128 + 255) / 256, 256, 0, stream>>>(agg, cnt, N);
    count_kernel<<<(E + 255) / 256, 256, 0, stream>>>(colI, cnt, E);

    const int nodeBlocks = (N / 16 + EW - 1) / EW;   // 782
    const int edgeBlocks = (E / 16 + EW - 1) / EW;   // 9375

    // Encoders
    enc_kernel<<<nodeBlocks, 128, 0, stream>>>(x, 8, ne_W1, ne_W2,
                                               F(118), F(119), F(121), F(120),
                                               h, nullptr, N);
    enc_kernel<<<edgeBlocks, 128, 0, stream>>>(eattr, 4, ee_W1, ee_W2,
                                               F(16), F(17), F(19), F(18),
                                               nullptr, ea, E);
    // Message-passing layers
    for (int i = 0; i < 8; ++i) {
        const int b = 20 + i * 12;
        edge_mp_kernel<<<edgeBlocks, 128, 0, stream>>>(
            h, ea, agg, rowI, colI, leW1[i], leW2[i],
            F(b + 2), F(b + 3), F(b + 5), F(b + 4), E);
        node_mp_kernel<<<nodeBlocks, 128, 0, stream>>>(
            h, agg, cnt, lnW1[i], lnW2[i],
            F(b + 8), F(b + 9), F(b + 11), F(b + 10), N);
    }
    // Decoders
    dec_kernel<<<nodeBlocks, 128, 0, stream>>>(
        h, dP_W1, dP_W2, F(11), F(12), F(10), F(13),
        dU_W1, dU_W2, F(5), F(6), F(4), F(7), out, N);
}
